// CausalSelfAttention_88682484728041
// MI455X (gfx1250) — matrix operand order repaired and checked by reading, not compiled, not run on hardware
//
#include <hip/hip_runtime.h>

typedef __attribute__((ext_vector_type(8)))  _Float16 v8h;
typedef __attribute__((ext_vector_type(16))) _Float16 v16h;
typedef __attribute__((ext_vector_type(8)))  float    v8f;

// ---------------------------------------------------------------------------
// Problem constants (from reference): B=2, T=2048, C=1024, H=16, Dh=64
// ---------------------------------------------------------------------------
#define TT   2048
#define CC   1024
#define HH   16
#define DH   64
#define BT   4096        // B*T
#define N3C  3072        // 3*C

#define BM   64          // block tile M
#define BN   256         // block tile N (8 waves: 2 in M x 4 in N, 32x64 each)
#define LDP  40          // padded LDS row stride (halves) -> 80B, 16B-aligned

static __device__ inline v8f zero8() {
  v8f z;
#pragma unroll
  for (int i = 0; i < 8; ++i) z[i] = 0.0f;
  return z;
}

// WMMA wrapper: D = A(16x32 f16) * B(32x16 f16) + C(16x16 f32)
static __device__ inline v8f wmma32(v16h a, v16h b, v8f c) {
  return __builtin_amdgcn_wmma_f32_16x16x32_f16(
      /*neg_a=*/false, a, /*neg_b=*/false, b,
      /*c_mod=*/(short)0, c, /*reuse_a=*/false, /*reuse_b=*/false);
}

// A-fragment loader: p points at tile (row0,col0) of row-major f16 [.., ld]
// lane L<16: row=L, K elems {0..15}; lane L>=16: row=L-16, {16..31}
static __device__ inline v16h ldA(const _Float16* p, int ld) {
  const int lane = threadIdx.x & 31;
  const int row  = lane & 15;
  const int cb   = (lane & 16) ? 16 : 0;
  const _Float16* q = p + row * ld + cb;
  v8h lo = *(const v8h*)(q);
  v8h hi = *(const v8h*)(q + 8);
  v16h a;
#pragma unroll
  for (int i = 0; i < 8; ++i) { a[i] = lo[i]; a[i + 8] = hi[i]; }
  return a;
}

// B-fragment loader: B is 32(K) x 16(N); memory holds, for each column n,
// K contiguous at p + n*ldk. lane L: n=L&15, K base = (L&16)?16:0 -> one 32B load.
static __device__ inline v16h ldB(const _Float16* p, int ldk) {
  const int lane = threadIdx.x & 31;
  const int n    = lane & 15;
  const int kb   = (lane & 16) ? 16 : 0;
  return *(const v16h*)(p + (size_t)n * ldk + kb);
}

// ---------------------------------------------------------------------------
// CDNA5 async global->LDS copy (16B per lane) + explicit ASYNCcnt wait.
// Inline asm keeps this portable across ROCm-7.2 / amdgpu-toolchain builtins.
// ---------------------------------------------------------------------------
static __device__ inline void async_copy16(const _Float16* g, _Float16* l) {
  unsigned loff =
      (unsigned)(unsigned long long)(__attribute__((address_space(3))) _Float16*)l;
  unsigned long long ga = (unsigned long long)g;
  asm volatile("global_load_async_to_lds_b128 %0, %1, off"
               :: "v"(loff), "v"(ga) : "memory");
}
static __device__ inline void wait_async0() {
  asm volatile("s_wait_asynccnt 0x0" ::: "memory");
}

// ---------------------------------------------------------------------------
// Conversion / transpose kernels (fp32 -> f16)
// ---------------------------------------------------------------------------
__global__ void cvt_f16_kernel(const float* __restrict__ in,
                               _Float16* __restrict__ out, int n) {
  int i = blockIdx.x * blockDim.x + threadIdx.x;
  if (i < n) out[i] = (_Float16)in[i];
}

// in: [K,N] row-major fp32 ; out: [N,K] row-major f16
__global__ void cvt_transpose_kernel(const float* __restrict__ in,
                                     _Float16* __restrict__ out, int K, int N) {
  int i = blockIdx.x * blockDim.x + threadIdx.x;
  if (i < K * N) {
    int n = i / K, k = i - n * K;
    out[(size_t)n * K + k] = (_Float16)in[(size_t)k * N + n];
  }
}

// ---------------------------------------------------------------------------
// GEMM1: qkv = x @ Wqkv + bqkv, split-scatter epilogue into
//   Q  [B,H,T,Dh] f16 (pre-scaled by 1/sqrt(Dh)), K [B,H,T,Dh], VT [B,H,Dh,T]
// Block 256 thr = 8 waves (2M x 4N), block tile 64x256, wave tile 32x64.
// A tile (64x32/k-step) double-buffered in LDS via async global->LDS loads.
// ---------------------------------------------------------------------------
__global__ void gemm_qkv_kernel(const _Float16* __restrict__ X,   // [BT, C]
                                const _Float16* __restrict__ Wt,  // [3C, C]
                                const float*    __restrict__ bias,// [3C]
                                _Float16* __restrict__ Q,
                                _Float16* __restrict__ Kd,
                                _Float16* __restrict__ VT) {
  __shared__ _Float16 Abuf[2][BM * LDP];
  const int lane = threadIdx.x & 31;
  const int wave = threadIdx.x >> 5;
  const int wm = wave >> 2;                 // 0..1
  const int wn = wave & 3;                  // 0..3
  const int m0 = blockIdx.x * BM;
  const int n0 = blockIdx.y * BN + wn * 64;

  // async-stage mapping: thread -> (row, 16B segment) of the 64x32 A tile
  const int arow = threadIdx.x >> 2;        // 0..63
  const int aseg = threadIdx.x & 3;         // 0..3  (4 x 16B = 64B row)
  const _Float16* Ag = X + (size_t)(m0 + arow) * CC + aseg * 8;
  const int aoff = arow * LDP + aseg * 8;

  v8f acc[2][4];
#pragma unroll
  for (int mi = 0; mi < 2; ++mi)
#pragma unroll
    for (int j = 0; j < 4; ++j) acc[mi][j] = zero8();

  async_copy16(Ag, &Abuf[0][aoff]);         // k = 0
  wait_async0();
  __syncthreads();

  for (int ki = 0; ki < CC / 32; ++ki) {
    const int k = ki * 32;
    const int cur = ki & 1;
    if (ki + 1 < CC / 32)
      async_copy16(Ag + k + 32, &Abuf[cur ^ 1][aoff]);

    v16h b[4];
#pragma unroll
    for (int j = 0; j < 4; ++j)
      b[j] = ldB(Wt + (size_t)(n0 + j * 16) * CC + k, CC);
    v16h a0 = ldA(&Abuf[cur][(wm * 32 +  0) * LDP], LDP);
    v16h a1 = ldA(&Abuf[cur][(wm * 32 + 16) * LDP], LDP);
#pragma unroll
    for (int j = 0; j < 4; ++j) {
      acc[0][j] = wmma32(a0, b[j], acc[0][j]);
      acc[1][j] = wmma32(a1, b[j], acc[1][j]);
    }
    wait_async0();                          // prefetch landed in LDS
    __syncthreads();                        // whole tile visible to all waves
  }

  // Epilogue: wave tile 64-wide / 64-aligned -> single region (q/k/v) + head
  const int region  = n0 >> 10;             // 0=q, 1=k, 2=v
  const int h       = (n0 & 1023) >> 6;
  const int rowHalf = (lane & 16) ? 8 : 0;
#pragma unroll
  for (int mi = 0; mi < 2; ++mi) {
#pragma unroll
    for (int j = 0; j < 4; ++j) {
      const int n  = n0 + j * 16 + (lane & 15);
      const int dh = n & 63;
      const float bv = bias[n];
#pragma unroll
      for (int r = 0; r < 8; ++r) {
        const int row = m0 + wm * 32 + mi * 16 + r + rowHalf;
        const int b_ = row >> 11;           // / T
        const int t  = row & (TT - 1);
        const float val = acc[mi][j][r] + bv;
        const int bh = b_ * HH + h;
        if (region == 0) {
          Q[((size_t)bh * TT + t) * DH + dh] = (_Float16)(val * 0.125f);
        } else if (region == 1) {
          Kd[((size_t)bh * TT + t) * DH + dh] = (_Float16)val;
        } else {
          VT[((size_t)bh * DH + dh) * TT + t] = (_Float16)val;
        }
      }
    }
  }
}

// ---------------------------------------------------------------------------
// Flash attention: one wave owns one (b,h, 16-row q-tile).
// S = Q K^T (pre-scaled), online softmax, O += P V. Y: [B,T,H*Dh] f16.
// block = 256 (8 waves), grid = B*H*(T/16)/8 = 512
// ---------------------------------------------------------------------------
__global__ void attn_kernel(const _Float16* __restrict__ Q,
                            const _Float16* __restrict__ Kd,
                            const _Float16* __restrict__ VT,
                            _Float16* __restrict__ Y) {
  __shared__ _Float16 lds[8][16 * LDP];      // per-wave 16x32 P tile, padded
  const int lane = threadIdx.x & 31;
  const int wave = threadIdx.x >> 5;
  const int idx  = blockIdx.x * 8 + wave;    // [0, B*H*T/16)
  const int qt = idx & 127;                  // q tile within (b,h)
  const int bh = idx >> 7;                   // b*H + h
  const int q0 = qt * 16;

  const _Float16* Qb = Q  + (size_t)bh * TT * DH;
  const _Float16* Kb = Kd + (size_t)bh * TT * DH;
  const _Float16* Vb = VT + (size_t)bh * DH * TT;

  // Q fragments for the whole tile (16 x 64), kept in registers
  v16h qa0 = ldA(Qb + (size_t)q0 * DH + 0,  DH);
  v16h qa1 = ldA(Qb + (size_t)q0 * DH + 32, DH);

  v8f o[4];
#pragma unroll
  for (int j = 0; j < 4; ++j) o[j] = zero8();
  float m[8], l[8];
#pragma unroll
  for (int r = 0; r < 8; ++r) { m[r] = -1e30f; l[r] = 0.0f; }

  const int rowHalf = (lane & 16) ? 8 : 0;
  _Float16* P = &lds[wave][0];

  for (int kb = 0; kb < q0 + 16; kb += 32) {
    // ---- S tile: 16 q-rows x 32 keys, f32 ----
    v8f s0 = zero8(), s1 = zero8();
    {
      v16h b00 = ldB(Kb + (size_t)(kb +  0) * DH + 0,  DH);
      v16h b01 = ldB(Kb + (size_t)(kb +  0) * DH + 32, DH);
      v16h b10 = ldB(Kb + (size_t)(kb + 16) * DH + 0,  DH);
      v16h b11 = ldB(Kb + (size_t)(kb + 16) * DH + 32, DH);
      s0 = wmma32(qa0, b00, s0);
      s0 = wmma32(qa1, b01, s0);
      s1 = wmma32(qa0, b10, s1);
      s1 = wmma32(qa1, b11, s1);
    }

    // ---- causal mask (only the diagonal chunk needs it) ----
    if (kb + 31 > q0) {
      const int key0 = kb + (lane & 15);
      const int key1 = kb + 16 + (lane & 15);
#pragma unroll
      for (int r = 0; r < 8; ++r) {
        const int row = q0 + r + rowHalf;
        if (key0 > row) s0[r] = -1e30f;
        if (key1 > row) s1[r] = -1e30f;
      }
    }

    // ---- online softmax (row stats via half-wave shfl reductions) ----
#pragma unroll
    for (int r = 0; r < 8; ++r) {
      float mx = fmaxf(s0[r], s1[r]);
      mx = fmaxf(mx, __shfl_xor(mx, 1));
      mx = fmaxf(mx, __shfl_xor(mx, 2));
      mx = fmaxf(mx, __shfl_xor(mx, 4));
      mx = fmaxf(mx, __shfl_xor(mx, 8));
      const float mn = fmaxf(m[r], mx);
      const float alpha = __expf(m[r] - mn);
      m[r] = mn;
      const float p0 = __expf(s0[r] - mn);
      const float p1 = __expf(s1[r] - mn);
      s0[r] = p0; s1[r] = p1;
      float rs = p0 + p1;
      rs += __shfl_xor(rs, 1);
      rs += __shfl_xor(rs, 2);
      rs += __shfl_xor(rs, 4);
      rs += __shfl_xor(rs, 8);
      l[r] = l[r] * alpha + rs;
#pragma unroll
      for (int j = 0; j < 4; ++j) o[j][r] *= alpha;
    }

    // ---- D-layout -> A-layout for P through per-wave LDS tile ----
#pragma unroll
    for (int r = 0; r < 8; ++r) {
      P[(r + rowHalf) * LDP + (lane & 15)]      = (_Float16)s0[r];
      P[(r + rowHalf) * LDP + 16 + (lane & 15)] = (_Float16)s1[r];
    }
    asm volatile("" ::: "memory");           // order LDS stores before reloads
    v16h pa = ldA(P, LDP);

    // ---- O += P @ V  (VT gives contiguous B-fragments) ----
#pragma unroll
    for (int j = 0; j < 4; ++j) {
      v16h b = ldB(Vb + (size_t)(j * 16) * TT + kb, TT);
      o[j] = wmma32(pa, b, o[j]);
    }
  }

  // ---- epilogue: Y[b, t, h*64 + dh] = O / l ----
  const int b = bh >> 4, h = bh & 15;
#pragma unroll
  for (int j = 0; j < 4; ++j) {
    const int dh = j * 16 + (lane & 15);
#pragma unroll
    for (int r = 0; r < 8; ++r) {
      const int t = q0 + r + rowHalf;
      Y[((size_t)(b * TT + t)) * CC + h * DH + dh] = (_Float16)(o[j][r] / l[r]);
    }
  }
}

// ---------------------------------------------------------------------------
// GEMM2: out = y @ Wproj + bproj (fp32 out). Same blocking as GEMM1.
// ---------------------------------------------------------------------------
__global__ void gemm_proj_kernel(const _Float16* __restrict__ Yh,  // [BT, C]
                                 const _Float16* __restrict__ Wt,  // [C, C]^T
                                 const float*    __restrict__ bias,
                                 float* __restrict__ out) {
  __shared__ _Float16 Abuf[2][BM * LDP];
  const int lane = threadIdx.x & 31;
  const int wave = threadIdx.x >> 5;
  const int wm = wave >> 2;
  const int wn = wave & 3;
  const int m0 = blockIdx.x * BM;
  const int n0 = blockIdx.y * BN + wn * 64;

  const int arow = threadIdx.x >> 2;
  const int aseg = threadIdx.x & 3;
  const _Float16* Ag = Yh + (size_t)(m0 + arow) * CC + aseg * 8;
  const int aoff = arow * LDP + aseg * 8;

  v8f acc[2][4];
#pragma unroll
  for (int mi = 0; mi < 2; ++mi)
#pragma unroll
    for (int j = 0; j < 4; ++j) acc[mi][j] = zero8();

  async_copy16(Ag, &Abuf[0][aoff]);
  wait_async0();
  __syncthreads();

  for (int ki = 0; ki < CC / 32; ++ki) {
    const int k = ki * 32;
    const int cur = ki & 1;
    if (ki + 1 < CC / 32)
      async_copy16(Ag + k + 32, &Abuf[cur ^ 1][aoff]);

    v16h b[4];
#pragma unroll
    for (int j = 0; j < 4; ++j)
      b[j] = ldB(Wt + (size_t)(n0 + j * 16) * CC + k, CC);
    v16h a0 = ldA(&Abuf[cur][(wm * 32 +  0) * LDP], LDP);
    v16h a1 = ldA(&Abuf[cur][(wm * 32 + 16) * LDP], LDP);
#pragma unroll
    for (int j = 0; j < 4; ++j) {
      acc[0][j] = wmma32(a0, b[j], acc[0][j]);
      acc[1][j] = wmma32(a1, b[j], acc[1][j]);
    }
    wait_async0();
    __syncthreads();
  }

  const int rowHalf = (lane & 16) ? 8 : 0;
#pragma unroll
  for (int mi = 0; mi < 2; ++mi) {
#pragma unroll
    for (int j = 0; j < 4; ++j) {
      const int n = n0 + j * 16 + (lane & 15);
      const float bv = bias[n];
#pragma unroll
      for (int r = 0; r < 8; ++r) {
        const int row = m0 + wm * 32 + mi * 16 + r + rowHalf;
        out[(size_t)row * CC + n] = acc[mi][j][r] + bv;
      }
    }
  }
}

// ---------------------------------------------------------------------------
// Launch
// ---------------------------------------------------------------------------
extern "C" void kernel_launch(void* const* d_in, const int* in_sizes, int n_in,
                              void* d_out, int out_size, void* d_ws, size_t ws_size,
                              hipStream_t stream) {
  const float* x     = (const float*)d_in[0];   // [2,2048,1024]
  const float* Wqkv  = (const float*)d_in[1];   // [1024,3072]
  const float* bqkv  = (const float*)d_in[2];   // [3072]
  const float* Wproj = (const float*)d_in[3];   // [1024,1024]
  const float* bproj = (const float*)d_in[4];   // [1024]
  float* out = (float*)d_out;

  const size_t MB = 1ull << 20;
  char* ws = (char*)d_ws;
  _Float16* xh  = (_Float16*)(ws +  0 * MB);  // 8 MiB  : x as f16 [BT,C]
  _Float16* WqT = (_Float16*)(ws +  8 * MB);  // 6 MiB  : Wqkv^T f16 [3C,C]
  _Float16* WpT = (_Float16*)(ws + 14 * MB);  // 2 MiB  : Wproj^T f16 [C,C]
  _Float16* Qb  = (_Float16*)(ws + 16 * MB);  // 8 MiB  : Q [B,H,T,Dh] (scaled)
  _Float16* Kb  = (_Float16*)(ws + 24 * MB);  // 8 MiB  : K [B,H,T,Dh]
  _Float16* VTb = (_Float16*)(ws + 32 * MB);  // 8 MiB  : V^T [B,H,Dh,T]
  _Float16* Yh  = (_Float16*)(ws + 40 * MB);  // 8 MiB  : attn out [BT,C]

  // fp32 -> f16 conversions (+ weight transposes for contiguous B-fragments)
  cvt_f16_kernel<<<(BT * CC) / 256, 256, 0, stream>>>(x, xh, BT * CC);
  cvt_transpose_kernel<<<(CC * N3C) / 256, 256, 0, stream>>>(Wqkv, WqT, CC, N3C);
  cvt_transpose_kernel<<<(CC * CC) / 256, 256, 0, stream>>>(Wproj, WpT, CC, CC);

  // qkv = x @ Wqkv + b, scattered into Q / K / V^T
  dim3 g1(BT / BM, N3C / BN);
  gemm_qkv_kernel<<<g1, 256, 0, stream>>>(xh, WqT, bqkv, Qb, Kb, VTb);

  // flash attention, 16-row q-tile per wave
  attn_kernel<<<(2 * HH * (TT / 16)) / 8, 256, 0, stream>>>(Qb, Kb, VTb, Yh);

  // out = y @ Wproj + b (fp32)
  dim3 g2(BT / BM, CC / BN);
  gemm_proj_kernel<<<g2, 256, 0, stream>>>(Yh, WpT, bproj, out);
}
